// DCWM_43069932044713
// MI455X (gfx1250) — matrix operand order repaired and checked by reading, not compiled, not run on hardware
//
#include <hip/hip_runtime.h>
#include <cstddef>

// ---------------------------------------------------------------------------
// Problem constants (from the reference)
// ---------------------------------------------------------------------------
#define T_STEPS 128
#define IN_DIM  256
#define IFACE   350
#define CTRL    512
#define OUT_DIM 128
#define NNET    5000
#define XIN_DIM (IN_DIM + IFACE)   // 606
#define Y_DIM   (IFACE + OUT_DIM)  // 478

// Padded matrix geometry for WMMA tiling:
//   columns padded to 160*32 = 5120  -> each of the 32 waves owns exactly
//                                       5 contiguous K-chunks (uniform trip)
//   rows    padded to 313*16 = 5008
#define NPADC   5120
#define NROWP   5008
#define NBLK    313
#define KCHUNKS 160
#define CPW     5            // chunks per wave (160 / 32)
#define FORGET  (1.0f / 3.0f)

typedef __attribute__((ext_vector_type(16))) __bf16         v16bf;
typedef __attribute__((ext_vector_type(8)))  float          v8f;
typedef __attribute__((ext_vector_type(8)))  unsigned short ushort8;

union BfPack { ushort8 h[2]; v16bf v; };

__device__ __forceinline__ unsigned short f2bf(float f) {
    unsigned int u = __float_as_uint(f);
    u += 0x7FFFu + ((u >> 16) & 1u);     // round-to-nearest-even
    return (unsigned short)(u >> 16);
}
__device__ __forceinline__ float bf2f(unsigned short h) {
    return __uint_as_float(((unsigned int)h) << 16);
}

// WGP-scope prefetch (scope 0): pulls into all cache levels on miss, stops on
// WGP-cache hit.  The __builtin_prefetch path only emits SE/DEV scope, which
// skips the WGP cache entirely -- useless here since the matrix is already
// L2-resident.  Three prefetches cover one lane's 320B next-block segment.
__device__ __forceinline__ void prefetch_wgp_320(const void* p) {
    unsigned long long a = (unsigned long long)p;
    asm volatile("global_prefetch_b8 %0, off\n\t"
                 "global_prefetch_b8 %0, off offset:128\n\t"
                 "global_prefetch_b8 %0, off offset:256"
                 :: "v"(a));
}

// ---------------------------------------------------------------------------
// Prep kernel 1: fp32 randomnetwork -> zero-padded bf16 matrix (EXACT:
// entries are {-1, 0.5, 0}, all representable in bf16).
// ---------------------------------------------------------------------------
__global__ void prep_mat_kernel(const float* __restrict__ rn,
                                unsigned short* __restrict__ matPad) {
    size_t stride = (size_t)gridDim.x * blockDim.x;
    size_t total  = (size_t)NROWP * NPADC;
    for (size_t idx = (size_t)blockIdx.x * blockDim.x + threadIdx.x;
         idx < total; idx += stride) {
        int r = (int)(idx / NPADC);
        int c = (int)(idx % NPADC);
        float v = (r < NNET && c < NNET) ? rn[(size_t)r * NNET + c] : 0.0f;
        matPad[idx] = f2bf(v);
    }
}

// ---------------------------------------------------------------------------
// Prep kernel 2: generic fp32 GEMM C[M,N] = A[M,K] @ B[K,N] (+ bias[n]).
// One-time collapse of the affine MLP chain: Wc = W1@W2@W3@W4,
// bc = ((b1@W2+b2)@W3+b3)@W4 + b4.  (~0.9 GFLOP, grid-parallel.)
// ---------------------------------------------------------------------------
__global__ void gemm_f32_kernel(const float* __restrict__ A,
                                const float* __restrict__ B,
                                const float* __restrict__ bias,
                                float* __restrict__ C,
                                int M, int N, int K) {
    int idx = blockIdx.x * blockDim.x + threadIdx.x;
    if (idx >= M * N) return;
    int m = idx / N, n = idx % N;
    float acc = bias ? bias[n] : 0.0f;
    for (int k = 0; k < K; ++k) acc = fmaf(A[m * K + k], B[k * N + n], acc);
    C[idx] = acc;
}

// ---------------------------------------------------------------------------
// Main persistent kernel: one workgroup, 1024 threads = 32 waves.
// The whole 128-step recurrence runs here. The serial Gauss-Seidel chain is
// synchronized with cheap on-WGP barriers (2 per 16-row block), state lives
// in LDS as fp32 plus a bf16 hi/lo split (matrix entries are exact in bf16,
// so dot(row,hi)+dot(row,lo) gives ~fp32-accurate dot products), and each
// block's 16x5120 GEMV runs as two independent fully-unrolled chains of 5
// v_wmma_f32_16x16x32_bf16 per wave sharing the same A tiles.
// ---------------------------------------------------------------------------
__global__ void __launch_bounds__(1024)
recurrent_net_kernel(const float* __restrict__ x,
                     const float* __restrict__ iface0,
                     const unsigned short* __restrict__ mat,
                     const float* __restrict__ Wc,
                     const float* __restrict__ bc,
                     float* __restrict__ out) {
    __shared__ float                        s_state[NPADC];      // fp32 state
    __shared__ __align__(16) unsigned short s_state_hi[NPADC];   // bf16(state)
    __shared__ __align__(16) unsigned short s_state_lo[NPADC];   // bf16(state - hi)
    __shared__ float                        s_xin[XIN_DIM];
    __shared__ float                        s_iface[IFACE];
    __shared__ float                        s_part[32 * 16];     // per-wave partials
    __shared__ float                        s_diag[16 * 16];     // diagonal block
    __shared__ float                        s_blk[16];

    const int tid  = threadIdx.x;
    const int lane = tid & 31;   // wave32
    const int wv   = tid >> 5;   // wave id 0..31
    const int mrow = lane & 15;  // WMMA M row handled by this lane
    const int hi   = lane >> 4;  // lane group: selects K sub-ranges per ISA layout
    const int c0   = wv * CPW;   // this wave's first K-chunk (contiguous run)

    // Init: state = 0, iface = interface0
    for (int i = tid; i < NPADC; i += 1024) {
        s_state[i] = 0.0f; s_state_hi[i] = 0; s_state_lo[i] = 0;
    }
    for (int i = tid; i < IFACE; i += 1024) s_iface[i] = iface0[i];
    __syncthreads();

    for (int t = 0; t < T_STEPS; ++t) {
        // ---- collapsed MLP: y = sigmoid(x_in @ Wc + bc) ----
        for (int i = tid; i < XIN_DIM; i += 1024)
            s_xin[i] = (i < IN_DIM) ? x[t * IN_DIM + i] : s_iface[i - IN_DIM];
        __syncthreads();

        if (tid < Y_DIM) {
            float acc = bc[tid];
            for (int j = 0; j < XIN_DIM; ++j)
                acc = fmaf(s_xin[j], Wc[j * Y_DIM + tid], acc);
            float yv = 1.0f / (1.0f + __expf(-acc));
            if (tid < OUT_DIM) {
                out[t * OUT_DIM + tid] = yv;             // out_v = y[:128]
            } else {
                int k = tid - OUT_DIM;                   // state[:350] += y[128:]
                float nf = s_state[k] + yv;
                s_state[k] = nf;
                unsigned short h = f2bf(nf);
                s_state_hi[k] = h;
                s_state_lo[k] = f2bf(nf - bf2f(h));
            }
        }
        __syncthreads();

        // ---- blocked Gauss-Seidel neuron scan ----
        for (int blk = 0; blk < NBLK; ++blk) {
            const int rowBase = blk * 16;

            // Per-lane pointers: A tile rows from L2-resident bf16 matrix.
            const unsigned short* arow =
                mat + (size_t)(rowBase + mrow) * NPADC + c0 * 32;
            const int sb = c0 * 32 + hi * 8;

            // Stage next block's row segment into the WGP cache while this
            // block's WMMA chain runs.
            if (blk + 1 < NBLK)
                prefetch_wgp_320(arow + (size_t)16 * NPADC);

            // base_i = dot(row_i, state_snapshot): new values for j < rowBase,
            // old for j >= rowBase (diagonal block fixed up below).
            // Two chains share the A tile: hi-state and lo-state residual.
            v8f accH = {0.f, 0.f, 0.f, 0.f, 0.f, 0.f, 0.f, 0.f};
            v8f accL = {0.f, 0.f, 0.f, 0.f, 0.f, 0.f, 0.f, 0.f};
            #pragma unroll
            for (int u = 0; u < CPW; ++u) {
                BfPack A, Bh, Bl;
                // 16-bit A 16x32 layout: lanes 0-15 hold K=0..7,16..23;
                // lanes 16-31 hold K=8..15,24..31 (ISA 7.12.2)
                A.h[0] = *(const ushort8*)(arow + u * 32 + hi * 8);
                A.h[1] = *(const ushort8*)(arow + u * 32 + 16 + hi * 8);
                // B 32x16: broadcast state chunk to all 16 columns;
                // lanes 0-15 hold K=0..7,16..23, lanes 16-31 hold K=8..15,24..31
                Bh.h[0] = *(const ushort8*)(&s_state_hi[sb + u * 32]);
                Bh.h[1] = *(const ushort8*)(&s_state_hi[sb + u * 32 + 16]);
                Bl.h[0] = *(const ushort8*)(&s_state_lo[sb + u * 32]);
                Bl.h[1] = *(const ushort8*)(&s_state_lo[sb + u * 32 + 16]);
                accH = __builtin_amdgcn_wmma_f32_16x16x32_bf16(
                           false, A.v, false, Bh.v, (short)0, accH, false, false);
                accL = __builtin_amdgcn_wmma_f32_16x16x32_bf16(
                           false, A.v, false, Bl.v, (short)0, accL, false, false);
            }

            // Every C column is identical; lane 0 holds rows 0-7, lane 16 rows 8-15.
            if (lane == 0) {
                #pragma unroll
                for (int r = 0; r < 8; ++r)
                    s_part[wv * 16 + r] = accH[r] + accL[r];
            } else if (lane == 16) {
                #pragma unroll
                for (int r = 0; r < 8; ++r)
                    s_part[wv * 16 + 8 + r] = accH[r] + accL[r];
            } else if (lane >= 1 && lane <= 8) {
                // stage the 16x16 diagonal block into LDS for the serial fixup
                int di = wv * 8 + (lane - 1);           // 0..255
                s_diag[di] = bf2f(mat[(size_t)(rowBase + (di >> 4)) * NPADC
                                      + rowBase + (di & 15)]);
            }
            __syncthreads();

            // Wave 0 does reduction AND the serial fixup: intra-wave LDS
            // ordering (in-order DScnt) makes a workgroup barrier between the
            // two unnecessary.
            if (wv == 0) {
                if (lane < 16) {
                    float s = 0.0f;
                    for (int w = 0; w < 32; ++w) s += s_part[w * 16 + lane];
                    s_blk[lane] = s * FORGET;
                }
                // Sequential in-block Gauss-Seidel fixup:
                // s_i += FORGET * sum_{j<i} R[i,j] * (new_j - old_j)
                if (lane == 0) {
                    float delta[16];
                    #pragma unroll 1
                    for (int i2 = 0; i2 < 16; ++i2) {
                        float s = s_blk[i2];
                        for (int j = 0; j < i2; ++j)
                            s = fmaf(FORGET * s_diag[i2 * 16 + j], delta[j], s);
                        float ns = fmaxf(s, 0.0f);
                        int gi = rowBase + i2;
                        delta[i2]   = ns - s_state[gi];
                        s_state[gi] = ns;
                        unsigned short h = f2bf(ns);
                        s_state_hi[gi] = h;
                        s_state_lo[gi] = f2bf(ns - bf2f(h));
                    }
                }
            }
            __syncthreads();
        }

        // iface = state[:350] after the scan
        for (int i = tid; i < IFACE; i += 1024) s_iface[i] = s_state[i];
        __syncthreads();
    }
}

// ---------------------------------------------------------------------------
// Launcher
// ---------------------------------------------------------------------------
extern "C" void kernel_launch(void* const* d_in, const int* in_sizes, int n_in,
                              void* d_out, int out_size, void* d_ws, size_t ws_size,
                              hipStream_t stream) {
    const float* x      = (const float*)d_in[0];   // [128,256]
    const float* iface0 = (const float*)d_in[1];   // [350]
    const float* rn     = (const float*)d_in[2];   // [5000,5000]
    const float* W1     = (const float*)d_in[3];   // [606,512]
    const float* b1     = (const float*)d_in[4];
    const float* W2     = (const float*)d_in[5];   // [512,1024]
    const float* b2     = (const float*)d_in[6];
    const float* W3     = (const float*)d_in[7];   // [1024,512]
    const float* b3     = (const float*)d_in[8];
    const float* W4     = (const float*)d_in[9];   // [512,478]
    const float* b4     = (const float*)d_in[10];
    float* out = (float*)d_out;                    // [128,128]

    // Workspace layout (~56.2 MB total)
    char* ws = (char*)d_ws;
    auto align16 = [](size_t v) { return (v + 15) & ~(size_t)15; };
    size_t off = 0;
    unsigned short* matPad = (unsigned short*)(ws + off);
    off = align16(off + (size_t)NROWP * NPADC * sizeof(unsigned short)); // 51.3 MB
    float* T1 = (float*)(ws + off); off = align16(off + (size_t)XIN_DIM * 1024 * 4);
    float* T2 = (float*)(ws + off); off = align16(off + (size_t)XIN_DIM * 512 * 4);
    float* WcP = (float*)(ws + off); off = align16(off + (size_t)XIN_DIM * Y_DIM * 4);
    float* bt1 = (float*)(ws + off); off = align16(off + 1024 * 4);
    float* bt2 = (float*)(ws + off); off = align16(off + 512 * 4);
    float* bcP = (float*)(ws + off); off = align16(off + Y_DIM * 4);

    // 1) lossless bf16 conversion of the random network (padded for WMMA tiles)
    prep_mat_kernel<<<4096, 256, 0, stream>>>(rn, matPad);

    // 2) collapse the affine MLP chain offline: Wc = W1@W2@W3@W4
    {
        int n;
        n = XIN_DIM * 1024;
        gemm_f32_kernel<<<(n + 255) / 256, 256, 0, stream>>>(W1, W2, nullptr, T1, XIN_DIM, 1024, 512);
        n = XIN_DIM * 512;
        gemm_f32_kernel<<<(n + 255) / 256, 256, 0, stream>>>(T1, W3, nullptr, T2, XIN_DIM, 512, 1024);
        n = XIN_DIM * Y_DIM;
        gemm_f32_kernel<<<(n + 255) / 256, 256, 0, stream>>>(T2, W4, nullptr, WcP, XIN_DIM, Y_DIM, 512);
        // bias chain: bc = ((b1@W2+b2)@W3+b3)@W4 + b4
        gemm_f32_kernel<<<4, 256, 0, stream>>>(b1, W2, b2, bt1, 1, 1024, 512);
        gemm_f32_kernel<<<2, 256, 0, stream>>>(bt1, W3, b3, bt2, 1, 512, 1024);
        gemm_f32_kernel<<<2, 256, 0, stream>>>(bt2, W4, b4, bcP, 1, Y_DIM, 512);
    }

    // 3) the whole 128-step recurrence in one persistent workgroup
    recurrent_net_kernel<<<1, 1024, 0, stream>>>(x, iface0, matPad, WcP, bcP, out);
}
